// APPNP_65549790871810
// MI455X (gfx1250) — matrix operand order repaired and checked by reading, not compiled, not run on hardware
//
#include <hip/hip_runtime.h>

typedef __attribute__((ext_vector_type(16))) _Float16 v16h;
typedef __attribute__((ext_vector_type(8)))  _Float16 v8h;
typedef __attribute__((ext_vector_type(8)))  float    v8f;

#define IN_DIM 512
#define OUT_DIM 64
#define K_STEPS 10
#define ALPHA_C 0.1f

// ---------------- utility kernels ----------------

__global__ void zero_kernel(float* __restrict__ p, unsigned n) {
    unsigned i = blockIdx.x * 256u + threadIdx.x;
    if (i < n) p[i] = 0.f;
}

// norms[0..N)   accumulates deg_out (indexed by src)
// norms[N..2N)  accumulates deg_in  (indexed by dst)
__global__ void degree_kernel(const int* __restrict__ src, const int* __restrict__ dst,
                              float* __restrict__ norms, int N, int E) {
    int e = blockIdx.x * 256 + threadIdx.x;
    if (e < E) {
        atomicAdd(&norms[src[e]], 1.f);
        atomicAdd(&norms[N + dst[e]], 1.f);
    }
}

__global__ void rsqrt_kernel(float* __restrict__ norms, int n) {
    int i = blockIdx.x * 256 + threadIdx.x;
    if (i < n) norms[i] = rsqrtf(fmaxf(norms[i], 1.f));
}

// Pre-swizzle W1 [512,64] and W2 [64,64] (f32, row-major) into f16 WMMA
// B-fragment order: flat index = ((kt*4 + nt)*32 + lane)*16 + e
//   col = nt*16 + (lane&15);  k = kt*32 + ((lane>>4)<<3) + (e&7) + ((e>>3)<<4)
__global__ void convert_weights_kernel(const float* __restrict__ W1, const float* __restrict__ W2,
                                       _Float16* __restrict__ W1sw, _Float16* __restrict__ W2sw) {
    int tid = threadIdx.x;
    for (int i = tid; i < 16 * 4 * 32 * 16; i += 256) {
        int e = i & 15, lane = (i >> 4) & 31, nt = (i >> 9) & 3, kt = i >> 11;
        int col = nt * 16 + (lane & 15);
        int k = kt * 32 + ((lane >> 4) << 3) + (e & 7) + ((e >> 3) << 4);
        W1sw[i] = (_Float16)W1[k * OUT_DIM + col];
    }
    for (int i = tid; i < 2 * 4 * 32 * 16; i += 256) {
        int e = i & 15, lane = (i >> 4) & 31, nt = (i >> 9) & 3, kt = i >> 11;
        int col = nt * 16 + (lane & 15);
        int k = kt * 32 + ((lane >> 4) << 3) + (e & 7) + ((e >> 3) << 4);
        W2sw[i] = (_Float16)W2[k * OUT_DIM + col];
    }
}

// ---------------- GEMM1: h1 = relu(X @ W1 + b1), X f32 [N,512] -> h1h f16 [N,64] ----------------
// 256 threads = 8 waves; each wave computes a 16x64 tile via 4 N-tiles of v_wmma_f32_16x16x32_f16.
__global__ void __launch_bounds__(256)
gemm1_kernel(const float* __restrict__ X, const _Float16* __restrict__ W1sw,
             const float* __restrict__ b1, _Float16* __restrict__ h1h, int nrows) {
    __shared__ v16h sB[16 * 4 * 32];                 // 2048 fragments-halves groups = 64 KB
    {
        const int4* gsrc = (const int4*)W1sw;
        int4* ldst = (int4*)sB;
        for (int i = threadIdx.x; i < (16 * 4 * 32 * 16) / 8; i += 256) ldst[i] = gsrc[i];
    }
    __syncthreads();

    int tid = threadIdx.x;
    int wid = tid >> 5, lane = tid & 31;
    int rowBase = (blockIdx.x * 8 + wid) * 16;
    if (rowBase >= nrows) return;                    // whole-wave guard: EXEC stays all-ones

    int row  = rowBase + (lane & 15);
    int kOff = (lane >> 4) << 3;                     // 0 or 8
    const float* xrow = X + (size_t)row * IN_DIM;

    v8f acc[4] = {v8f{}, v8f{}, v8f{}, v8f{}};

    for (int kt = 0; kt < 16; ++kt) {
        const float* ap = xrow + kt * 32 + kOff;
        float4 f0 = *(const float4*)(ap);
        float4 f1 = *(const float4*)(ap + 4);
        float4 f2 = *(const float4*)(ap + 16);
        float4 f3 = *(const float4*)(ap + 20);
        v16h a;
        a[0] = (_Float16)f0.x;  a[1] = (_Float16)f0.y;  a[2]  = (_Float16)f0.z;  a[3]  = (_Float16)f0.w;
        a[4] = (_Float16)f1.x;  a[5] = (_Float16)f1.y;  a[6]  = (_Float16)f1.z;  a[7]  = (_Float16)f1.w;
        a[8] = (_Float16)f2.x;  a[9] = (_Float16)f2.y;  a[10] = (_Float16)f2.z;  a[11] = (_Float16)f2.w;
        a[12] = (_Float16)f3.x; a[13] = (_Float16)f3.y; a[14] = (_Float16)f3.z;  a[15] = (_Float16)f3.w;

#pragma unroll
        for (int nt = 0; nt < 4; ++nt) {
            v16h b = sB[(kt * 4 + nt) * 32 + lane];
            acc[nt] = __builtin_amdgcn_wmma_f32_16x16x32_f16(
                false, a, false, b, (short)0, acc[nt], false, false);
        }
    }

    int colBase = lane & 15;
    int rowOff  = (lane >> 4) << 3;
#pragma unroll
    for (int nt = 0; nt < 4; ++nt) {
        int col = nt * 16 + colBase;
        float bias = b1[col];
#pragma unroll
        for (int r = 0; r < 8; ++r) {
            int m = rowBase + rowOff + r;
            float v = acc[nt][r] + bias;
            v = v > 0.f ? v : 0.f;
            h1h[(size_t)m * OUT_DIM + col] = (_Float16)v;
        }
    }
}

// ---------------- GEMM2: h0 = h1 @ W2 + b2 ; also hs = h0 * norm_src ----------------
__global__ void __launch_bounds__(256)
gemm2_kernel(const _Float16* __restrict__ h1h, const _Float16* __restrict__ W2sw,
             const float* __restrict__ b2, const float* __restrict__ norms,
             float* __restrict__ h0, float* __restrict__ hs, int nrows) {
    __shared__ v16h sB[2 * 4 * 32];                  // 8 KB
    {
        const int4* gsrc = (const int4*)W2sw;
        int4* ldst = (int4*)sB;
        for (int i = threadIdx.x; i < (2 * 4 * 32 * 16) / 8; i += 256) ldst[i] = gsrc[i];
    }
    __syncthreads();

    int tid = threadIdx.x;
    int wid = tid >> 5, lane = tid & 31;
    int rowBase = (blockIdx.x * 8 + wid) * 16;
    if (rowBase >= nrows) return;

    int row  = rowBase + (lane & 15);
    int kOff = (lane >> 4) << 3;

    v8f acc[4] = {v8f{}, v8f{}, v8f{}, v8f{}};

#pragma unroll
    for (int kt = 0; kt < 2; ++kt) {
        const _Float16* ap = h1h + (size_t)row * OUT_DIM + kt * 32 + kOff;
        v8h lo = *(const v8h*)(ap);
        v8h hi = *(const v8h*)(ap + 16);
        v16h a;
#pragma unroll
        for (int i = 0; i < 8; ++i) { a[i] = lo[i]; a[i + 8] = hi[i]; }

#pragma unroll
        for (int nt = 0; nt < 4; ++nt) {
            v16h b = sB[(kt * 4 + nt) * 32 + lane];
            acc[nt] = __builtin_amdgcn_wmma_f32_16x16x32_f16(
                false, a, false, b, (short)0, acc[nt], false, false);
        }
    }

    int colBase = lane & 15;
    int rowOff  = (lane >> 4) << 3;
#pragma unroll
    for (int nt = 0; nt < 4; ++nt) {
        int col = nt * 16 + colBase;
        float bias = b2[col];
#pragma unroll
        for (int r = 0; r < 8; ++r) {
            int m = rowBase + rowOff + r;
            float v = acc[nt][r] + bias;
            size_t idx = (size_t)m * OUT_DIM + col;
            h0[idx] = v;
            hs[idx] = v * norms[m];                  // norm_src
        }
    }
}

// ---------------- propagation ----------------
// agg[dst,:] += hs[src,:]   (64 threads per edge, 1 float each; h fits in 192MB L2)
__global__ void scatter_kernel(const int* __restrict__ src, const int* __restrict__ dst,
                               const float* __restrict__ hs, float* __restrict__ agg,
                               unsigned total) {
    unsigned gid = blockIdx.x * 256u + threadIdx.x;
    if (gid >= total) return;
    unsigned e = gid >> 6;
    unsigned f = gid & 63u;
    int s = src[e], d = dst[e];
    atomicAdd(&agg[(size_t)d * OUT_DIM + f], hs[(size_t)s * OUT_DIM + f]);
}

// h = 0.9*agg*norm_dst + 0.1*h0 ; out = h ; hs = h*norm_src ; agg reset to 0
__global__ void combine_kernel(float* __restrict__ agg, const float* __restrict__ norms,
                               const float* __restrict__ h0, float* __restrict__ hs,
                               float* __restrict__ out, int N, unsigned total) {
    unsigned gid = blockIdx.x * 256u + threadIdx.x;
    if (gid >= total) return;
    int row = gid >> 6;
    float a = agg[gid];
    agg[gid] = 0.f;
    float h = (1.f - ALPHA_C) * a * norms[N + row] + ALPHA_C * h0[gid];
    out[gid] = h;
    hs[gid] = h * norms[row];
}

// ---------------- launcher ----------------

extern "C" void kernel_launch(void* const* d_in, const int* in_sizes, int n_in,
                              void* d_out, int out_size, void* d_ws, size_t ws_size,
                              hipStream_t stream) {
    const float* X  = (const float*)d_in[0];
    const int* edge = (const int*)d_in[1];
    const float* W1 = (const float*)d_in[2];
    const float* b1 = (const float*)d_in[3];
    const float* W2 = (const float*)d_in[4];
    const float* b2 = (const float*)d_in[5];

    int N = in_sizes[0] / IN_DIM;
    int E = in_sizes[1] / 2;
    const int* src = edge;
    const int* dst = edge + E;

    // workspace carve (f32 regions first, then f16; all 16B-aligned)
    float* norms = (float*)d_ws;                     // 2N   (norm_src | norm_dst)
    float* agg   = norms + 2 * (size_t)N;            // 64N
    float* h0    = agg + 64 * (size_t)N;             // 64N
    float* hs    = h0 + 64 * (size_t)N;              // 64N
    _Float16* h1h  = (_Float16*)(hs + 64 * (size_t)N); // 64N halves
    _Float16* W1sw = h1h + 64 * (size_t)N;           // 32768 halves
    _Float16* W2sw = W1sw + 16 * 4 * 32 * 16;        // 4096 halves
    float* out = (float*)d_out;

    // zero norms + agg (contiguous 66N floats)
    unsigned zcnt = (unsigned)(66 * (size_t)N);
    zero_kernel<<<(zcnt + 255) / 256, 256, 0, stream>>>(norms, zcnt);

    degree_kernel<<<(E + 255) / 256, 256, 0, stream>>>(src, dst, norms, N, E);
    rsqrt_kernel<<<(2 * N + 255) / 256, 256, 0, stream>>>(norms, 2 * N);
    convert_weights_kernel<<<1, 256, 0, stream>>>(W1, W2, W1sw, W2sw);

    int gblocks = (N + 127) / 128;                   // 8 waves x 16 rows per block
    gemm1_kernel<<<gblocks, 256, 0, stream>>>(X, W1sw, b1, h1h, N);
    gemm2_kernel<<<gblocks, 256, 0, stream>>>(h1h, W2sw, b2, norms, h0, hs, N);

    unsigned scnt = (unsigned)((size_t)E * OUT_DIM);
    unsigned ccnt = (unsigned)((size_t)N * OUT_DIM);
    for (int k = 0; k < K_STEPS; ++k) {
        scatter_kernel<<<(scnt + 255) / 256, 256, 0, stream>>>(src, dst, hs, agg, scnt);
        combine_kernel<<<(ccnt + 255) / 256, 256, 0, stream>>>(agg, norms, h0, hs, out, N, ccnt);
    }
}
